// _NLM_Layer_67087389164126
// MI455X (gfx1250) — matrix operand order repaired and checked by reading, not compiled, not run on hardware
//
#include <hip/hip_runtime.h>
#include <hip/hip_bf16.h>

typedef __attribute__((ext_vector_type(16))) _Float16 v16h_t;
typedef __attribute__((ext_vector_type(8)))  _Float16 v8h_t;
typedef __attribute__((ext_vector_type(8)))  float    v8f_t;

// Fast sigmoid: v_exp_f32 + v_rcp_f32 (avoids the IEEE divide expansion).
__device__ __forceinline__ float fast_sigmoid(float x) {
    return __builtin_amdgcn_rcpf(1.f + __expf(-x));
}

// ---------------------------------------------------------------------------
// f32 -> f16 conversion
// ---------------------------------------------------------------------------
__global__ void cvt_f32_f16(const float* __restrict__ s, _Float16* __restrict__ d, int n) {
    int stride = gridDim.x * blockDim.x;
    for (int i = blockIdx.x * blockDim.x + threadIdx.x; i < n; i += stride)
        d[i] = (_Float16)s[i];
}

// ---------------------------------------------------------------------------
// r2x/r2n : masked max/min over j of X2[b,i,j,f]  (mask m2 : j != i)
//   max includes a 0 candidate (masked slot), min includes a 1 candidate.
// ---------------------------------------------------------------------------
__global__ void reduce2_kernel(const float* __restrict__ X2,
                               _Float16* __restrict__ rx, _Float16* __restrict__ rn) {
    int idx = blockIdx.x * 256 + threadIdx.x;           // [0, 16*32*16)
    int f = idx & 15, i = (idx >> 4) & 31, b = idx >> 9;
    const float* base = X2 + ((size_t)(((b << 5) | i) << 5) << 4) + f;
    float mx = 0.f, mn = 1.f;
    for (int j = 0; j < 32; ++j) {
        if (j != i) {
            float v = base[j << 4];
            mx = fmaxf(mx, v);
            mn = fminf(mn, v);
        }
    }
    rx[idx] = (_Float16)mx;
    rn[idx] = (_Float16)mn;
}

// ---------------------------------------------------------------------------
// r3x/r3n : masked max/min over k of X3[b,i,j,k,f] (mask m3 : i,j,k distinct)
// ---------------------------------------------------------------------------
__global__ void reduce3_kernel(const float* __restrict__ X3,
                               _Float16* __restrict__ rx, _Float16* __restrict__ rn) {
    int idx = blockIdx.x * 256 + threadIdx.x;           // [0, 16*32*32*16)
    int f = idx & 15, j = (idx >> 4) & 31, i = (idx >> 9) & 31, b = idx >> 14;
    float mx = 0.f, mn = 1.f;
    if (i != j) {
        const float* base = X3 + ((size_t)(((((b << 5) | i) << 5) | j) << 5) << 4) + f;
        for (int k = 0; k < 32; ++k) {
            if (k != i && k != j) {
                float v = base[k << 4];
                mx = fmaxf(mx, v);
                mn = fminf(mn, v);
            }
        }
    }
    rx[idx] = (_Float16)mx;
    rn[idx] = (_Float16)mn;
}

// ---------------------------------------------------------------------------
// Pack a KxN f32 weight matrix into WMMA B-fragment order (f16).
// Fragment (kt,nt) is 32x16; lane L holds column N = nt*16+(L&15),
// halves h = K = kt*32 + (h>>3)*16 + (L>>4)*8 + (h&7)   (table order).
// ---------------------------------------------------------------------------
__global__ void pack_w_kernel(const float* __restrict__ w, _Float16* __restrict__ p,
                              int K, int N) {
    int n = K * N;
    int nnt = N >> 4;
    int stride = gridDim.x * blockDim.x;
    for (int e = blockIdx.x * blockDim.x + threadIdx.x; e < n; e += stride) {
        int fi = e >> 9, rem = e & 511;
        int lane = rem >> 4, h = rem & 15;
        int kt = fi / nnt, nt = fi - kt * nnt;
        int krow = kt * 32 + ((h >> 3) << 4) + ((lane >> 4) << 3) + (h & 7);
        int ncol = (nt << 4) + (lane & 15);
        p[e] = (_Float16)w[krow * N + ncol];
    }
}

// ---------------------------------------------------------------------------
// Arity-0 branch: c0 = [X0, max_n X1, min_n X1] (16 rows x 48) -> MLP -> out0.
// ---------------------------------------------------------------------------
__global__ void __launch_bounds__(256) c0_kernel(
    const float* __restrict__ X0, const float* __restrict__ X1,
    const float* __restrict__ W1, const float* __restrict__ b1,
    const float* __restrict__ W2, const float* __restrict__ b2,
    float* __restrict__ out0) {
    __shared__ float sC[16 * 48];
    __shared__ float sH[16 * 64];
    int t = threadIdx.x;
    {   // phase 1: reductions + concat
        int b = t >> 4, f = t & 15;
        float mx = -1e30f, mn = 1e30f;
        const float* base = X1 + ((b << 5) << 4) + f;
        for (int n2 = 0; n2 < 32; ++n2) {
            float v = base[n2 << 4];
            mx = fmaxf(mx, v);
            mn = fminf(mn, v);
        }
        sC[b * 48 + f]      = X0[(b << 4) + f];
        sC[b * 48 + 16 + f] = mx;
        sC[b * 48 + 32 + f] = mn;
    }
    __syncthreads();
    for (int p = t; p < 16 * 64; p += 256) {   // phase 2: hidden
        int b = p >> 6, h = p & 63;
        float s = b1[h];
        for (int d2 = 0; d2 < 48; ++d2) s += sC[b * 48 + d2] * W1[d2 * 64 + h];
        sH[(b << 6) + h] = fast_sigmoid(s);
    }
    __syncthreads();
    {   // phase 3: output + passthrough
        int b = t >> 4, o = t & 15;
        float s = b2[o];
        for (int h = 0; h < 64; ++h) s += sH[(b << 6) + h] * W2[(h << 4) + o];
        out0[(b << 5) + o]      = fast_sigmoid(s);
        out0[(b << 5) + 16 + o] = X0[(b << 4) + o];
    }
}

// ---------------------------------------------------------------------------
// Passthrough: out[row,16:32] = X[row,:]
// ---------------------------------------------------------------------------
__global__ void copy_pass_kernel(const float* __restrict__ X, float* __restrict__ outseg, int n) {
    int stride = gridDim.x * blockDim.x;
    for (int idx = blockIdx.x * blockDim.x + threadIdx.x; idx < n; idx += stride) {
        int row = idx >> 4, f = idx & 15;
        outseg[(size_t)row * 32 + 16 + f] = X[idx];
    }
}

// ---------------------------------------------------------------------------
// Main WMMA MLP kernel. MODE 1: c1 (K=64), MODE 2: c2 (K=128), MODE 3: c3 (K=192).
// 4 waves/block; each wave owns one 16-row tile at a time.
// ---------------------------------------------------------------------------
template<int MODE>
__global__ void __launch_bounds__(128) mlp_wmma_kernel(
    const _Float16* __restrict__ X0h, const _Float16* __restrict__ X1h,
    const _Float16* __restrict__ X2h, const _Float16* __restrict__ X3h,
    const _Float16* __restrict__ r2x, const _Float16* __restrict__ r2n,
    const _Float16* __restrict__ r3x, const _Float16* __restrict__ r3n,
    const _Float16* __restrict__ W1p, const _Float16* __restrict__ W2p,
    const float* __restrict__ b1, const float* __restrict__ b2,
    float* __restrict__ outseg, int ntiles, int tstride) {

    constexpr int KCHUNKS = (MODE == 1) ? 4 : ((MODE == 2) ? 8 : 12);
    constexpr int KT      = KCHUNKS / 2;          // number of 32-wide K tiles
    constexpr int PER     = KCHUNKS / 2;          // chunks staged per lane
    constexpr int ASTRIDE = KCHUNKS * 16 + 8;     // halves; keeps 16B alignment
    constexpr int NW1     = KT * 4 * 512;         // packed W1 halves

    __shared__ _Float16 sW1[NW1];
    __shared__ _Float16 sW2[2 * 512];
    __shared__ _Float16 sA[4][16 * ASTRIDE];
    __shared__ _Float16 sH[4][16 * 72];

    int tid = threadIdx.x;
    for (int e = tid; e < NW1 / 8; e += 128)
        ((uint4*)sW1)[e] = ((const uint4*)W1p)[e];
    for (int e = tid; e < 1024 / 8; e += 128)
        ((uint4*)sW2)[e] = ((const uint4*)W2p)[e];
    __syncthreads();

    int wave = tid >> 5, lane = tid & 31;
    int arow = lane & 15;             // A-fragment: row carried by this lane
    int akb  = (lane >> 4) << 3;      // A-fragment: K sub-base (0 or 8)
    int dcol = lane & 15;             // C/D fragment: column
    int drow = (lane >> 4) << 3;      // C/D fragment: row base (0 or 8)

    // staging: lane pair (2m, 2m+1) owns row m; even lane stages chunks
    // [0,PER), odd lane [PER,2*PER)  -- no integer division anywhere.
    int smrow = lane >> 1;
    int scbase = (lane & 1) * PER;

    // hoist loop-invariant biases out of the tile loop
    float bb1[4];
    #pragma unroll
    for (int nt = 0; nt < 4; ++nt) bb1[nt] = b1[(nt << 4) + dcol];
    float bb2 = b2[dcol];

    for (int t = blockIdx.x * 4 + wave; t < ntiles; t += tstride) {
        int row0 = t << 4;
        _Float16* A = &sA[wave][0];

        if (MODE == 3) {
            // warm L2 with next tile's contiguous X3 block (512B)
            int tn = t + tstride;
            if (tn < ntiles && lane < 4)
                __builtin_prefetch(X3h + ((size_t)tn << 8) + (lane << 6), 0, 1);
        }

        // ---- stage 16 rows x KCHUNKS chunks (16 halves each) into LDS ----
        #pragma unroll
        for (int it = 0; it < PER; ++it) {
            int c = scbase + it;
            int row = row0 + smrow;
            const _Float16* src;
            if (MODE == 1) {
                int b = row >> 5;
                src = (c == 0) ? X0h + (size_t)b * 16
                    : (c == 1) ? X1h + (size_t)row * 16
                    : (c == 2) ? r2x + (size_t)row * 16
                               : r2n + (size_t)row * 16;
            } else if (MODE == 2) {
                int b = row >> 10, i = (row >> 5) & 31, j = row & 31;
                int bji = (b << 10) | (j << 5) | i;
                switch (c) {
                    case 0:  src = X1h + (size_t)(row >> 5) * 16; break;
                    case 1:  src = X2h + (size_t)row * 16; break;
                    case 2:  src = r3x + (size_t)row * 16; break;
                    case 3:  src = r3n + (size_t)row * 16; break;
                    case 4:  src = X1h + (size_t)((b << 5) | j) * 16; break;
                    case 5:  src = X2h + (size_t)bji * 16; break;
                    case 6:  src = r3x + (size_t)bji * 16; break;
                    default: src = r3n + (size_t)bji * 16; break;
                }
            } else {
                int b = row >> 15, i = (row >> 10) & 31, j = (row >> 5) & 31, k = row & 31;
                switch (c) {   // permutations (i,j,k),(i,k,j),(j,i,k),(k,i,j),(j,k,i),(k,j,i)
                    case 0:  src = X2h + (size_t)((b << 10) | (i << 5) | j) * 16; break;
                    case 1:  src = X3h + (size_t)row * 16; break;
                    case 2:  src = X2h + (size_t)((b << 10) | (i << 5) | k) * 16; break;
                    case 3:  src = X3h + (size_t)((b << 15) | (i << 10) | (k << 5) | j) * 16; break;
                    case 4:  src = X2h + (size_t)((b << 10) | (j << 5) | i) * 16; break;
                    case 5:  src = X3h + (size_t)((b << 15) | (j << 10) | (i << 5) | k) * 16; break;
                    case 6:  src = X2h + (size_t)((b << 10) | (k << 5) | i) * 16; break;
                    case 7:  src = X3h + (size_t)((b << 15) | (k << 10) | (i << 5) | j) * 16; break;
                    case 8:  src = X2h + (size_t)((b << 10) | (j << 5) | k) * 16; break;
                    case 9:  src = X3h + (size_t)((b << 15) | (j << 10) | (k << 5) | i) * 16; break;
                    case 10: src = X2h + (size_t)((b << 10) | (k << 5) | j) * 16; break;
                    default: src = X3h + (size_t)((b << 15) | (k << 10) | (j << 5) | i) * 16; break;
                }
            }
            uint4* dst = (uint4*)(A + smrow * ASTRIDE + (c << 4));
            const uint4* s = (const uint4*)src;
            dst[0] = s[0];
            dst[1] = s[1];
        }
        // wave-private LDS tile; LDS ops are in-order within a wave -> no barrier

        // ---- layer 1: 16xK @ Kx64, f32 accumulate ----
        v8f_t acc[4] = {};
        #pragma unroll
        for (int kt = 0; kt < KT; ++kt) {
            const _Float16* ap = A + arow * ASTRIDE + (kt << 5) + akb;
            v8h_t alo = *(const v8h_t*)ap;
            v8h_t ahi = *(const v8h_t*)(ap + 16);
            v16h_t af;
            #pragma unroll
            for (int q = 0; q < 8; ++q) { af[q] = alo[q]; af[q + 8] = ahi[q]; }
            #pragma unroll
            for (int nt = 0; nt < 4; ++nt) {
                v16h_t bf = *(const v16h_t*)(sW1 + ((((kt << 2) | nt) * 32 + lane) << 4));
                acc[nt] = __builtin_amdgcn_wmma_f32_16x16x32_f16(
                    false, af, false, bf, (short)0, acc[nt], false, false);
            }
        }

        // ---- bias + sigmoid, reshuffle C-layout -> A-layout through LDS ----
        _Float16* H = &sH[wave][0];
        #pragma unroll
        for (int nt = 0; nt < 4; ++nt) {
            #pragma unroll
            for (int v = 0; v < 8; ++v) {
                float h = fast_sigmoid(acc[nt][v] + bb1[nt]);
                H[(drow + v) * 72 + (nt << 4) + dcol] = (_Float16)h;
            }
        }

        // ---- layer 2: 16x64 @ 64x16 ----
        v8f_t o = {};
        #pragma unroll
        for (int kt = 0; kt < 2; ++kt) {
            const _Float16* hp = H + arow * 72 + (kt << 5) + akb;
            v8h_t lo = *(const v8h_t*)hp;
            v8h_t hi = *(const v8h_t*)(hp + 16);
            v16h_t hf;
            #pragma unroll
            for (int q = 0; q < 8; ++q) { hf[q] = lo[q]; hf[q + 8] = hi[q]; }
            v16h_t wf = *(const v16h_t*)(sW2 + (((kt << 5) + lane) << 4));
            o = __builtin_amdgcn_wmma_f32_16x16x32_f16(
                false, hf, false, wf, (short)0, o, false, false);
        }

        // ---- bias + sigmoid + store y into out[row, 0:16] ----
        float* orow = outseg + (size_t)row0 * 32;
        #pragma unroll
        for (int v = 0; v < 8; ++v) {
            float y = fast_sigmoid(o[v] + bb2);
            orow[(size_t)(drow + v) * 32 + dcol] = y;
        }
    }
}

// ---------------------------------------------------------------------------
// Host launcher
// ---------------------------------------------------------------------------
extern "C" void kernel_launch(void* const* d_in, const int* in_sizes, int n_in,
                              void* d_out, int out_size, void* d_ws, size_t ws_size,
                              hipStream_t stream) {
    (void)in_sizes; (void)n_in; (void)out_size; (void)ws_size;

    const float* X0 = (const float*)d_in[0];
    const float* X1 = (const float*)d_in[1];
    const float* X2 = (const float*)d_in[2];
    const float* X3 = (const float*)d_in[3];
    const float* W1_[4], *b1_[4], *W2_[4], *b2_[4];
    for (int i = 0; i < 4; ++i) {
        W1_[i] = (const float*)d_in[4 + 4 * i];
        b1_[i] = (const float*)d_in[5 + 4 * i];
        W2_[i] = (const float*)d_in[6 + 4 * i];
        b2_[i] = (const float*)d_in[7 + 4 * i];
    }

    float* out  = (float*)d_out;
    float* out0 = out;
    float* out1 = out + 512;
    float* out2 = out1 + 16384;
    float* out3 = out2 + 524288;

    char* ws = (char*)d_ws;
    size_t off = 0;
    auto take = [&](size_t bytes) -> char* {
        char* p = ws + off;
        off += (bytes + 255) & ~(size_t)255;
        return p;
    };
    _Float16* X0h  = (_Float16*)take(256 * 2);
    _Float16* X1h  = (_Float16*)take(8192 * 2);
    _Float16* X2h  = (_Float16*)take(262144 * 2);
    _Float16* X3h  = (_Float16*)take((size_t)8388608 * 2);
    _Float16* r2xh = (_Float16*)take(8192 * 2);
    _Float16* r2nh = (_Float16*)take(8192 * 2);
    _Float16* r3xh = (_Float16*)take(262144 * 2);
    _Float16* r3nh = (_Float16*)take(262144 * 2);
    _Float16* W1p1 = (_Float16*)take(64 * 64 * 2);
    _Float16* W2p1 = (_Float16*)take(64 * 16 * 2);
    _Float16* W1p2 = (_Float16*)take(128 * 64 * 2);
    _Float16* W2p2 = (_Float16*)take(64 * 16 * 2);
    _Float16* W1p3 = (_Float16*)take(192 * 64 * 2);
    _Float16* W2p3 = (_Float16*)take(64 * 16 * 2);

    auto cvt = [&](const float* s, _Float16* d, int n) {
        int blocks = (n + 255) / 256;
        if (blocks > 2048) blocks = 2048;
        cvt_f32_f16<<<blocks, 256, 0, stream>>>(s, d, n);
    };
    cvt(X0, X0h, 256);
    cvt(X1, X1h, 8192);
    cvt(X2, X2h, 262144);
    cvt(X3, X3h, 8388608);

    reduce2_kernel<<<32, 256, 0, stream>>>(X2, r2xh, r2nh);
    reduce3_kernel<<<1024, 256, 0, stream>>>(X3, r3xh, r3nh);

    auto pack = [&](const float* w, _Float16* p, int K, int N) {
        int n = K * N;
        pack_w_kernel<<<(n + 255) / 256, 256, 0, stream>>>(w, p, K, N);
    };
    pack(W1_[1], W1p1, 64, 64);
    pack(W2_[1], W2p1, 64, 16);
    pack(W1_[2], W1p2, 128, 64);
    pack(W2_[2], W2p2, 64, 16);
    pack(W1_[3], W1p3, 192, 64);
    pack(W2_[3], W2p3, 64, 16);

    c0_kernel<<<1, 256, 0, stream>>>(X0, X1, W1_[0], b1_[0], W2_[0], b2_[0], out0);

    mlp_wmma_kernel<1><<<8, 128, 0, stream>>>(
        X0h, X1h, X2h, X3h, r2xh, r2nh, r3xh, r3nh,
        W1p1, W2p1, b1_[1], b2_[1], out1, 32, 8 * 4);
    mlp_wmma_kernel<2><<<256, 128, 0, stream>>>(
        X0h, X1h, X2h, X3h, r2xh, r2nh, r3xh, r3nh,
        W1p2, W2p2, b1_[2], b2_[2], out2, 1024, 256 * 4);
    mlp_wmma_kernel<3><<<2048, 128, 0, stream>>>(
        X0h, X1h, X2h, X3h, r2xh, r2nh, r3xh, r3nh,
        W1p3, W2p3, b1_[3], b2_[3], out3, 32768, 2048 * 4);

    auto cpy = [&](const float* X, float* seg, int n) {
        int blocks = (n + 255) / 256;
        if (blocks > 4096) blocks = 4096;
        copy_pass_kernel<<<blocks, 256, 0, stream>>>(X, seg, n);
    };
    cpy(X1, out1, 8192);
    cpy(X2, out2, 262144);
    cpy(X3, out3, 8388608);
}
